// CrossAttention_55533927137654
// MI455X (gfx1250) — matrix operand order repaired and checked by reading, not compiled, not run on hardware
//
#include <hip/hip_runtime.h>

// ---------------------------------------------------------------------------
// Problem constants (from reference)
// ---------------------------------------------------------------------------
#define BB 4
#define LQ 1024
#define LK 1024
#define DM 768
#define NH 12
#define HD 64
#define MROWS (BB * LQ)          // 4096
#define EPSV 1e-5f

typedef __attribute__((ext_vector_type(16))) _Float16 v16h;
typedef __attribute__((ext_vector_type(8)))  _Float16 v8h;
typedef __attribute__((ext_vector_type(8)))  float    v8f;
typedef __attribute__((ext_vector_type(4)))  int      v4i;

union HFrag { v16h v; v8h h[2]; };

// ---------------------------------------------------------------------------
// CDNA5 async global->LDS copy (ASYNCcnt-tracked), guarded so the file still
// compiles on toolchains without the gfx1250 async builtins.
// Builtin signature (probe round 2): (v4i addrspace(1)*, v4i addrspace(3)*,
// imm offset, imm cpol).
// ---------------------------------------------------------------------------
#if defined(__HIP_DEVICE_COMPILE__) && defined(__gfx1250__) && \
    __has_builtin(__builtin_amdgcn_global_load_async_to_lds_b128) && \
    __has_builtin(__builtin_amdgcn_s_wait_asynccnt)
#define USE_ASYNC_LDS 1
#else
#define USE_ASYNC_LDS 0
#endif

__device__ __forceinline__ void ld_async_b128(void* lds, const void* g) {
#if USE_ASYNC_LDS
  __builtin_amdgcn_global_load_async_to_lds_b128(
      (__attribute__((address_space(1))) v4i*)g,
      (__attribute__((address_space(3))) v4i*)lds, 0, 0);
#else
  *(v8h*)lds = *(const v8h*)g;
#endif
}

template <int N> __device__ __forceinline__ void wait_async() {
#if USE_ASYNC_LDS
  __builtin_amdgcn_s_wait_asynccnt((unsigned short)N);
#endif
}

// ---------------------------------------------------------------------------
// Utility: f32 -> f16 cast
// ---------------------------------------------------------------------------
__global__ __launch_bounds__(256) void k_cast_h(const float* __restrict__ in,
                                                _Float16* __restrict__ out, int n) {
  int i = blockIdx.x * 256 + threadIdx.x;
  if (i < n) out[i] = (_Float16)in[i];
}

// ---------------------------------------------------------------------------
// RMSNorm(q) * w_norm  -> f16   (one block per row of 768)
// ---------------------------------------------------------------------------
__global__ __launch_bounds__(256) void k_rmsnorm(const float* __restrict__ q,
                                                 const float* __restrict__ wn,
                                                 _Float16* __restrict__ out) {
  const int row = blockIdx.x;
  const float* x = q + (size_t)row * DM;
  int tid = threadIdx.x, lane = tid & 31, wave = tid >> 5;
  float s = 0.f;
  for (int i = tid; i < DM; i += 256) { float v = x[i]; s += v * v; }
  for (int off = 16; off; off >>= 1) s += __shfl_xor(s, off, 32);
  __shared__ float red[8];
  if (lane == 0) red[wave] = s;
  __syncthreads();
  if (wave == 0) {
    float t = (lane < 8) ? red[lane] : 0.f;
    for (int off = 4; off; off >>= 1) t += __shfl_xor(t, off, 32);
    if (lane == 0) red[0] = t;
  }
  __syncthreads();
  float rms = rsqrtf(red[0] * (1.0f / DM) + EPSV);
  for (int i = tid; i < DM; i += 256)
    out[(size_t)row * DM + i] = (_Float16)(x[i] * rms * wn[i]);
}

// ---------------------------------------------------------------------------
// Tiled WMMA GEMM:  C[M,N] = A[M,K]h * W[N,K]h^T + bias[N]
// block = 256 threads (8 waves), tile 64(M) x 128(N), K-step 32,
// double-buffered LDS with async global->LDS staging.
// ---------------------------------------------------------------------------
template <int NTOT, bool STORE_F32>
__global__ __launch_bounds__(256) void k_gemm_wmma(
    const _Float16* __restrict__ A, const _Float16* __restrict__ W,
    const float* __restrict__ bias, _Float16* __restrict__ Ch,
    float* __restrict__ Cf, int M, int K) {
  __shared__ _Float16 As[2][64][40];    // 32 + 8 pad (row stride 80B, 16B aligned)
  __shared__ _Float16 Ws[2][128][40];

  const int tid = threadIdx.x;
  const int wave = tid >> 5, lane = tid & 31;
  const int l16 = lane & 15, hs = lane >> 4;
  const int wm = wave & 3, wn = wave >> 2;
  const int m0 = blockIdx.x * 64, n0 = blockIdx.y * 128;

  // per-thread staging coordinates (3 x b128 per K-step)
  const int ar = tid >> 2, ac = (tid & 3) * 8;   // A: 64 x 32, 8 halves/thread
  const int wr = tid >> 1, wc = (tid & 1) * 16;  // W: 128 x 32, 16 halves/thread

  v8f acc[4] = {};

  const int T = K / 32;
  auto issue = [&](int t, int buf) {
    const int k0 = t * 32;
    ld_async_b128(&As[buf][ar][ac],     &A[(size_t)(m0 + ar) * K + k0 + ac]);
    ld_async_b128(&Ws[buf][wr][wc],     &W[(size_t)(n0 + wr) * K + k0 + wc]);
    ld_async_b128(&Ws[buf][wr][wc + 8], &W[(size_t)(n0 + wr) * K + k0 + wc + 8]);
  };

  issue(0, 0);
  for (int t = 0; t < T; ++t) {
    const int buf = t & 1;
    if (t + 1 < T) { issue(t + 1, buf ^ 1); wait_async<3>(); }
    else           { wait_async<0>(); }
    __syncthreads();

    // A fragment (16x32): lane l<16 holds row l, K {0..7,16..23}; l>=16: {8..15,24..31}
    HFrag a;
    {
      int r = wm * 16 + l16, cb = hs * 8;
      a.h[0] = *(const v8h*)&As[buf][r][cb];
      a.h[1] = *(const v8h*)&As[buf][r][cb + 16];
    }
#pragma unroll
    for (int s = 0; s < 4; ++s) {
      // B fragment (32x16): lane holds column (n), K {0..7,16..23} (lanes 0-15) / {8..15,24..31}
      HFrag b;
      int r = wn * 64 + s * 16 + l16, cb = hs * 8;
      b.h[0] = *(const v8h*)&Ws[buf][r][cb];
      b.h[1] = *(const v8h*)&Ws[buf][r][cb + 16];
      acc[s] = __builtin_amdgcn_wmma_f32_16x16x32_f16(
          false, a.v, false, b.v, (short)0, acc[s], false, false);
    }
    __syncthreads();
  }

#pragma unroll
  for (int s = 0; s < 4; ++s) {
#pragma unroll
    for (int j = 0; j < 8; ++j) {
      int row = m0 + wm * 16 + j + 8 * hs;
      int col = n0 + wn * 64 + s * 16 + l16;
      float v = acc[s][j] + bias[col];
      if constexpr (STORE_F32) Cf[(size_t)row * NTOT + col] = v;
      else                     Ch[(size_t)row * NTOT + col] = (_Float16)v;
    }
  }
}

// ---------------------------------------------------------------------------
// Polar bias: sum_{k,n} r^k (pb[k,n] cos(n t) + pc[k,n] sin(n t))
// cos t = dx/r, sin t = dy/r -> no atan2/cos/sin transcendentals; raw v_rsq.
// ---------------------------------------------------------------------------
__device__ __forceinline__ float polar_bias(float dx, float dy,
                                            const float* __restrict__ cb,
                                            const float* __restrict__ cc) {
  float r2 = dx * dx + dy * dy;
  bool nz = (r2 > 0.f);
  float inv = __builtin_amdgcn_rsqf(r2);   // raw v_rsq_f32
  inv = nz ? inv : 0.f;
  float r = r2 * inv;
  float c1 = nz ? dx * inv : 1.f;   // theta = atan2(0,0) = 0 -> cos=1
  float s1 = dy * inv;
  float bias = 0.f, cn = 1.f, sn = 0.f;
#pragma unroll
  for (int n = 0; n < 5; ++n) {
    float wb = cb[n] + cb[5 + n] * r + cb[10 + n] * r2;
    float wc = cc[n] + cc[5 + n] * r + cc[10 + n] * r2;
    bias += wb * cn + wc * sn;
    float cn2 = cn * c1 - sn * s1;
    sn = sn * c1 + cn * s1;
    cn = cn2;
  }
  return bias;
}

// ---------------------------------------------------------------------------
// Flash-attention with fused polar bias.
// grid = (LQ/128, H, B); block = 256 (8 waves, each owning 16 q-rows).
// K-step = 32 keys: S (2 x 16x16 f32) via 4 WMMAs, online softmax,
// P -> LDS (per-wave) -> A-layout, P*V via 4 WMMAs into 16x64 f32 acc.
// ---------------------------------------------------------------------------
__global__ __launch_bounds__(256) void k_attn_polar(
    const _Float16* __restrict__ Qp,   // [B*LQ][DM]  (head h at col h*HD)
    const _Float16* __restrict__ KVp,  // [B*LK][2*DM] (k at h*HD, v at DM+h*HD)
    const float* __restrict__ posq, const float* __restrict__ posk,
    const float* __restrict__ pb, const float* __restrict__ pc, // [H][3][5]
    _Float16* __restrict__ O) {        // [B*LQ][DM]
  const int qt = blockIdx.x, h = blockIdx.y, b = blockIdx.z;
  __shared__ _Float16 Ks[32][72];      // 32 keys x 64 hd (+8 pad)
  __shared__ _Float16 Vt[64][40];      // transposed: [hd][key] (+8 pad)
  __shared__ float pkx[32], pky[32];
  __shared__ _Float16 Ps[8][16][40];   // per-wave P scratch 16x32 (+8 pad)

  const int tid = threadIdx.x, wave = tid >> 5, lane = tid & 31;
  const int l16 = lane & 15, hs = lane >> 4;
  const int r0 = qt * 128 + wave * 16;  // q row base (within batch)
  const float scale = 0.125f;           // 1/sqrt(64)

  // Q A-fragments for hd 0-31 and 32-63 (held in registers for whole kernel)
  HFrag Aq0, Aq1;
  {
    const _Float16* qrow = Qp + ((size_t)(b * LQ + r0 + l16)) * DM + h * HD;
    int cb = hs * 8;
    Aq0.h[0] = *(const v8h*)&qrow[cb];
    Aq0.h[1] = *(const v8h*)&qrow[cb + 16];
    Aq1.h[0] = *(const v8h*)&qrow[32 + cb];
    Aq1.h[1] = *(const v8h*)&qrow[32 + cb + 16];
  }
  float pqx[8], pqy[8];
#pragma unroll
  for (int j = 0; j < 8; ++j) {
    int row = r0 + j + 8 * hs;
    pqx[j] = posq[((size_t)b * LQ + row) * 2 + 0];
    pqy[j] = posq[((size_t)b * LQ + row) * 2 + 1];
  }
  float cpb[15], cpc[15];
#pragma unroll
  for (int t = 0; t < 15; ++t) { cpb[t] = pb[h * 15 + t]; cpc[t] = pc[h * 15 + t]; }

  v8f oacc[4] = {};
  float mrow[8], lrow[8];
#pragma unroll
  for (int j = 0; j < 8; ++j) { mrow[j] = -3.0e38f; lrow[j] = 0.f; }

  const int tok = tid >> 3, vc = (tid & 7) * 8;
  for (int kt = 0; kt < LK; kt += 32) {
    __syncthreads();
    { // stage K tile (async to LDS), V tile transposed via registers
      const _Float16* krow =
          KVp + ((size_t)(b * LK + kt + tok)) * (2 * DM) + h * HD;
      ld_async_b128(&Ks[tok][vc], &krow[vc]);
      v8h vv = *(const v8h*)&krow[DM + vc];
#pragma unroll
      for (int i = 0; i < 8; ++i) Vt[vc + i][tok] = vv[i];
      if (kt + 32 < LK) {  // prefetch next K/V rows (global_prefetch_b8)
        const _Float16* nrow =
            KVp + ((size_t)(b * LK + kt + 32 + tok)) * (2 * DM) + h * HD;
        __builtin_prefetch(nrow + vc, 0, 1);
        __builtin_prefetch(nrow + DM + vc, 0, 1);
      }
    }
    if (tid < 32) {
      pkx[tid] = posk[((size_t)b * LK + kt + tid) * 2 + 0];
      pky[tid] = posk[((size_t)b * LK + kt + tid) * 2 + 1];
    }
    wait_async<0>();
    __syncthreads();

    // S = Q K^T for 32 keys -> two 16x16 accumulators
    v8f S0 = {}, S1 = {};
#pragma unroll
    for (int i = 0; i < 2; ++i) {
      HFrag bk;
      int kk = i * 16 + l16, cb = hs * 8;
      bk.h[0] = *(const v8h*)&Ks[kk][cb];
      bk.h[1] = *(const v8h*)&Ks[kk][cb + 16];
      v8f si = (i == 0) ? S0 : S1;
      si = __builtin_amdgcn_wmma_f32_16x16x32_f16(false, Aq0.v, false, bk.v,
                                                  (short)0, si, false, false);
      bk.h[0] = *(const v8h*)&Ks[kk][32 + cb];
      bk.h[1] = *(const v8h*)&Ks[kk][32 + cb + 16];
      si = __builtin_amdgcn_wmma_f32_16x16x32_f16(false, Aq1.v, false, bk.v,
                                                  (short)0, si, false, false);
      if (i == 0) S0 = si; else S1 = si;
    }

    // scale + fused polar bias
    float kx0 = pkx[l16], ky0 = pky[l16];
    float kx1 = pkx[16 + l16], ky1 = pky[16 + l16];
#pragma unroll
    for (int j = 0; j < 8; ++j) {
      S0[j] = S0[j] * scale + polar_bias(pqx[j] - kx0, pqy[j] - ky0, cpb, cpc);
      S1[j] = S1[j] * scale + polar_bias(pqx[j] - kx1, pqy[j] - ky1, cpb, cpc);
    }

    // online softmax (row reductions across 16-lane halves)
#pragma unroll
    for (int j = 0; j < 8; ++j) {
      float t = fmaxf(S0[j], S1[j]);
#pragma unroll
      for (int off = 1; off < 16; off <<= 1) t = fmaxf(t, __shfl_xor(t, off, 32));
      float mn = fmaxf(mrow[j], t);
      float alpha = __expf(mrow[j] - mn);
      mrow[j] = mn;
      float p0 = __expf(S0[j] - mn);
      float p1 = __expf(S1[j] - mn);
      float rs = p0 + p1;
#pragma unroll
      for (int off = 1; off < 16; off <<= 1) rs += __shfl_xor(rs, off, 32);
      lrow[j] = lrow[j] * alpha + rs;
      oacc[0][j] *= alpha; oacc[1][j] *= alpha;
      oacc[2][j] *= alpha; oacc[3][j] *= alpha;
      // C-layout -> row-major P scratch (f16)
      Ps[wave][j + 8 * hs][l16]      = (_Float16)p0;
      Ps[wave][j + 8 * hs][16 + l16] = (_Float16)p1;
    }
    __syncthreads();

    // P (16x32) as A-fragment; V^T columns as B-fragments; O += P*V
    HFrag Ap;
    {
      int cb = hs * 8;
      Ap.h[0] = *(const v8h*)&Ps[wave][l16][cb];
      Ap.h[1] = *(const v8h*)&Ps[wave][l16][cb + 16];
    }
#pragma unroll
    for (int s = 0; s < 4; ++s) {
      HFrag bv;
      int hd = s * 16 + l16, cb = hs * 8;
      bv.h[0] = *(const v8h*)&Vt[hd][cb];
      bv.h[1] = *(const v8h*)&Vt[hd][cb + 16];
      oacc[s] = __builtin_amdgcn_wmma_f32_16x16x32_f16(
          false, Ap.v, false, bv.v, (short)0, oacc[s], false, false);
    }
  }

  // normalize + store f16 to o_h [B*LQ][DM]
#pragma unroll
  for (int j = 0; j < 8; ++j) {
    float inv = 1.f / lrow[j];
    int row = r0 + j + 8 * hs;
#pragma unroll
    for (int s = 0; s < 4; ++s) {
      int hd = s * 16 + l16;
      O[((size_t)(b * LQ + row)) * DM + h * HD + hd] = (_Float16)(oacc[s][j] * inv);
    }
  }
}

// ---------------------------------------------------------------------------
// Host launcher
// ---------------------------------------------------------------------------
extern "C" void kernel_launch(void* const* d_in, const int* in_sizes, int n_in,
                              void* d_out, int out_size, void* d_ws, size_t ws_size,
                              hipStream_t stream) {
  const float* q      = (const float*)d_in[0];
  const float* kv     = (const float*)d_in[1];
  const float* posq   = (const float*)d_in[2];
  const float* posk   = (const float*)d_in[3];
  const float* w_norm = (const float*)d_in[4];
  const float* w_q    = (const float*)d_in[5];
  const float* b_q    = (const float*)d_in[6];
  const float* w_kv   = (const float*)d_in[7];
  const float* b_kv   = (const float*)d_in[8];
  const float* w_out  = (const float*)d_in[9];
  const float* b_out  = (const float*)d_in[10];
  const float* pb     = (const float*)d_in[11];
  const float* pc     = (const float*)d_in[12];
  float* out = (float*)d_out;

  // workspace carve-up (all f16)
  _Float16* base = (_Float16*)d_ws;
  const size_t SZ_MK  = (size_t)MROWS * DM;       // 3,145,728
  const size_t SZ_WQ  = (size_t)DM * DM;          //   589,824
  const size_t SZ_WKV = (size_t)2 * DM * DM;      // 1,179,648
  _Float16* qn_h   = base;                 base += SZ_MK;
  _Float16* kv_h   = base;                 base += SZ_MK;
  _Float16* wq_h   = base;                 base += SZ_WQ;
  _Float16* wkv_h  = base;                 base += SZ_WKV;
  _Float16* wout_h = base;                 base += SZ_WQ;
  _Float16* qp_h   = base;                 base += SZ_MK;
  _Float16* kvp_h  = base;                 base += 2 * SZ_MK;
  _Float16* o_h    = base;                 base += SZ_MK;

  // 1) casts
  k_cast_h<<<(int)((SZ_WQ  + 255) / 256), 256, 0, stream>>>(w_q,   wq_h,   (int)SZ_WQ);
  k_cast_h<<<(int)((SZ_WKV + 255) / 256), 256, 0, stream>>>(w_kv,  wkv_h,  (int)SZ_WKV);
  k_cast_h<<<(int)((SZ_WQ  + 255) / 256), 256, 0, stream>>>(w_out, wout_h, (int)SZ_WQ);
  k_cast_h<<<(int)((SZ_MK  + 255) / 256), 256, 0, stream>>>(kv,    kv_h,   (int)SZ_MK);

  // 2) RMSNorm(q)
  k_rmsnorm<<<MROWS, 256, 0, stream>>>(q, w_norm, qn_h);

  // 3) projections
  dim3 gq(MROWS / 64, DM / 128);
  k_gemm_wmma<DM, false><<<gq, 256, 0, stream>>>(qn_h, wq_h, b_q, qp_h, nullptr,
                                                 MROWS, DM);
  dim3 gkv(MROWS / 64, (2 * DM) / 128);
  k_gemm_wmma<2 * DM, false><<<gkv, 256, 0, stream>>>(kv_h, wkv_h, b_kv, kvp_h,
                                                      nullptr, MROWS, DM);

  // 4) fused flash attention with polar bias
  dim3 ga(LQ / 128, NH, BB);
  k_attn_polar<<<ga, 256, 0, stream>>>(qp_h, kvp_h, posq, posk, pb, pc, o_h);

  // 5) output projection (f32 out)
  dim3 go(MROWS / 64, DM / 128);
  k_gemm_wmma<DM, true><<<go, 256, 0, stream>>>(o_h, wout_h, b_out, nullptr, out,
                                                MROWS, DM);
}
